// MambaBlock_34694745817813
// MI455X (gfx1250) — matrix operand order repaired and checked by reading, not compiled, not run on hardware
//
#include <hip/hip_runtime.h>
#include <hip/hip_bf16.h>

#define D_MODEL 1024
#define D_INNER 2048
#define D_STATE 16
#define D_CONV  4
#define DT_RANK 64
#define B_SZ    2
#define L_SEQ   2048
#define M_ROWS  (B_SZ * L_SEQ)   // 4096 rows

typedef __bf16 bf16;
typedef __attribute__((ext_vector_type(16))) __bf16 v16bf;
typedef __attribute__((ext_vector_type(8)))  float  v8f;

__device__ __forceinline__ bf16 f2bf(float f) {
    unsigned u = __builtin_bit_cast(unsigned, f);
    unsigned r = (u + 0x7FFFu + ((u >> 16) & 1u)) >> 16;   // round-to-nearest-even
    unsigned short s = (unsigned short)r;
    return __builtin_bit_cast(bf16, s);
}

__device__ __forceinline__ float sigmoidf_(float x) { return 1.0f / (1.0f + expf(-x)); }

// ---------------------------------------------------------------- elementwise
__global__ void k_cvt_bf16(const float* __restrict__ src, bf16* __restrict__ dst, long n) {
    long i = (long)blockIdx.x * blockDim.x + threadIdx.x;
    if (i < n) dst[i] = f2bf(src[i]);
}

// W [R x C] f32 row-major -> Wt [C x R] bf16 row-major (B-fragments load contiguously)
__global__ void k_transpose_bf16(const float* __restrict__ W, bf16* __restrict__ Wt, int R, int C) {
    long i = (long)blockIdx.x * blockDim.x + threadIdx.x;
    long n = (long)R * C;
    if (i < n) {
        int r = (int)(i / C), c = (int)(i % C);
        Wt[(size_t)c * R + r] = f2bf(W[i]);
    }
}

// dt_low = x_dbl[:, 0:64] -> bf16 [M x 64]
__global__ void k_cvt_dtlow(const float* __restrict__ xdbl, bf16* __restrict__ dst) {
    long i = (long)blockIdx.x * blockDim.x + threadIdx.x;
    long n = (long)M_ROWS * DT_RANK;
    if (i < n) {
        int r = (int)(i / DT_RANK), c = (int)(i % DT_RANK);
        dst[i] = f2bf(xdbl[(size_t)r * 96 + c]);
    }
}

struct Frag32B { uint4 lo, hi; };

// ---------------------------------------------------------------- async global->LDS slab copy
// Copies a 128-row x 32-col bf16 slab (8 KB) with 512 16-byte async chunks, 256 threads.
__device__ __forceinline__ void async_slab(const bf16* __restrict__ g, int ldk,
                                           unsigned ldsbase, int tid) {
    #pragma unroll
    for (int j = 0; j < 2; ++j) {
        int id  = tid + 256 * j;              // 0..511
        int row = id >> 2;                    // 0..127
        int off = (id & 3) << 3;              // bf16 element offset: 0,8,16,24
        unsigned long long ga = (unsigned long long)(uintptr_t)(g + (size_t)row * ldk + off);
        unsigned la = ldsbase + (unsigned)((row * 32 + off) * 2);
        asm volatile("global_load_async_to_lds_b128 %0, %1, off"
                     :: "v"(la), "v"(ga) : "memory");
    }
}

__device__ __forceinline__ void wait_async_all() {
    asm volatile("s_wait_asynccnt 0x0" ::: "memory");
}

// ---------------------------------------------------------------- blocked WMMA GEMM
// C[M x N](f32) = A[M x K](bf16) * Bt[N x K]^T(bf16). Requires M%128==0, N%128==0, K%32==0.
// 256 threads = 8 waves; wave tile 64(M) x 32(N) -> block tile 128 x 128.
// Double-buffered LDS staging via GLOBAL_LOAD_ASYNC_TO_LDS_B128 (ASYNCcnt).
__global__ __launch_bounds__(256) void k_gemm_bf16_blk(const bf16* __restrict__ A,
                                                       const bf16* __restrict__ Bt,
                                                       float* __restrict__ C,
                                                       int Mr, int Nc, int K) {
    __shared__ bf16 sA[2][128 * 32];   // 2 x 8 KB
    __shared__ bf16 sB[2][128 * 32];   // 2 x 8 KB
    const int tid   = threadIdx.x;
    const int lane  = tid & 31;
    const int wave  = tid >> 5;
    const int wm    = wave >> 2;       // 0..1 : 64-row band
    const int wn    = wave & 3;        // 0..3 : 32-col band
    const int m16   = lane & 15;
    const int khalf = lane >> 4;
    const int m0    = blockIdx.y * 128;
    const int n0    = blockIdx.x * 128;

    const unsigned sA0 = (unsigned)(uintptr_t)&sA[0][0];
    const unsigned sA1 = (unsigned)(uintptr_t)&sA[1][0];
    const unsigned sB0 = (unsigned)(uintptr_t)&sB[0][0];
    const unsigned sB1 = (unsigned)(uintptr_t)&sB[1][0];

    v8f acc[4][2] = {};

    // prologue: stage k-slab 0 into buffer 0
    async_slab(A  + (size_t)m0 * K, K, sA0, tid);
    async_slab(Bt + (size_t)n0 * K, K, sB0, tid);
    wait_async_all();
    __syncthreads();

    int cur = 0;
    for (int kb = 0; kb < K; kb += 32) {
        if (kb + 32 < K) {   // overlap: stage next slab while computing current
            async_slab(A  + (size_t)m0 * K + kb + 32, K, cur ? sA0 : sA1, tid);
            async_slab(Bt + (size_t)n0 * K + kb + 32, K, cur ? sB0 : sB1, tid);
        }

        v16bf Af[4], Bf[2];
        #pragma unroll
        for (int am = 0; am < 4; ++am) {
            const bf16* p = &sA[cur][(wm * 64 + am * 16 + m16) * 32 + khalf * 8];
            Frag32B f; f.lo = *(const uint4*)p; f.hi = *(const uint4*)(p + 16);
            Af[am] = __builtin_bit_cast(v16bf, f);
        }
        #pragma unroll
        for (int bn = 0; bn < 2; ++bn) {
            const bf16* p = &sB[cur][(wn * 32 + bn * 16 + m16) * 32 + khalf * 8];
            Frag32B f; f.lo = *(const uint4*)p; f.hi = *(const uint4*)(p + 16);
            Bf[bn] = __builtin_bit_cast(v16bf, f);
        }
        #pragma unroll
        for (int am = 0; am < 4; ++am)
            #pragma unroll
            for (int bn = 0; bn < 2; ++bn)
                acc[am][bn] = __builtin_amdgcn_wmma_f32_16x16x32_bf16(
                    false, Af[am], false, Bf[bn], (short)0, acc[am][bn], false, false);

        wait_async_all();
        __syncthreads();
        cur ^= 1;
    }

    #pragma unroll
    for (int am = 0; am < 4; ++am) {
        const int r0 = m0 + wm * 64 + am * 16 + (khalf << 3);
        #pragma unroll
        for (int bn = 0; bn < 2; ++bn) {
            const int c0 = n0 + wn * 32 + bn * 16 + m16;
            #pragma unroll
            for (int r = 0; r < 8; ++r)
                C[(size_t)(r0 + r) * Nc + c0] = acc[am][bn][r];
        }
    }
}

// ---------------------------------------------------------------- simple WMMA GEMM (skinny N)
// One wave32 = one 16x16 tile; used where N is not a multiple of 128 (x_proj, N=96).
__global__ void k_gemm_bf16(const bf16* __restrict__ A, const bf16* __restrict__ Bt,
                            float* __restrict__ C, int Mr, int Nc, int K) {
    const int lane = threadIdx.x & 31;
    const int wave = threadIdx.x >> 5;
    const int tilesN = Nc >> 4;
    const int tilesTotal = (Mr >> 4) * tilesN;
    const int tile = blockIdx.x * 8 + wave;
    if (tile >= tilesTotal) return;               // wave-uniform: EXEC stays all-1s

    const int tm = tile / tilesN;
    const int tn = tile % tilesN;
    const int m16 = lane & 15;
    const int khalf = lane >> 4;

    const bf16* pa = A  + (size_t)((tm << 4) + m16) * K + khalf * 8;
    const bf16* pb = Bt + (size_t)((tn << 4) + m16) * K + khalf * 8;

    v8f acc = {};
    for (int kb = 0; kb < K; kb += 32) {
        Frag32B fa, fb;
        fa.lo = *(const uint4*)(pa + kb);
        fa.hi = *(const uint4*)(pa + kb + 16);
        fb.lo = *(const uint4*)(pb + kb);
        fb.hi = *(const uint4*)(pb + kb + 16);
        v16bf av = __builtin_bit_cast(v16bf, fa);
        v16bf bv = __builtin_bit_cast(v16bf, fb);
        acc = __builtin_amdgcn_wmma_f32_16x16x32_bf16(
            false, av, false, bv, (short)0, acc, false, false);
    }

    const int r0 = (tm << 4) + (khalf << 3);
    const int c0 = (tn << 4) + m16;
    #pragma unroll
    for (int r = 0; r < 8; ++r)
        C[(size_t)(r0 + r) * Nc + c0] = acc[r];
}

// ---------------------------------------------------------------- conv + SiLU
__global__ void k_conv_silu(const float* __restrict__ xr, const float* __restrict__ cw,
                            const float* __restrict__ cb, float* __restrict__ u32,
                            bf16* __restrict__ u16) {
    long i = (long)blockIdx.x * blockDim.x + threadIdx.x;
    if (i >= (long)M_ROWS * D_INNER) return;
    int d = (int)(i % D_INNER);
    int l = (int)((i / D_INNER) % L_SEQ);
    int b = (int)(i / ((long)D_INNER * L_SEQ));
    float s = cb[d];
    #pragma unroll
    for (int j = 0; j < D_CONV; ++j) {
        int ll = l - (D_CONV - 1) + j;
        if (ll >= 0)
            s += xr[((size_t)(b * L_SEQ + ll)) * (2 * D_INNER) + d] * cw[d * D_CONV + j];
    }
    float v = s * sigmoidf_(s);
    u32[i] = v;
    u16[i] = f2bf(v);
}

// ---------------------------------------------------------------- softplus(dt + b_dt)
__global__ void k_softplus(float* __restrict__ dt, const float* __restrict__ bdt) {
    long i = (long)blockIdx.x * blockDim.x + threadIdx.x;
    if (i >= (long)M_ROWS * D_INNER) return;
    int d = (int)(i % D_INNER);
    float v = dt[i] + bdt[d];
    dt[i] = (v > 20.0f) ? v : log1pf(expf(v));
}

// ---------------------------------------------------------------- selective scan
__global__ void k_scan(const float* __restrict__ dt, const float* __restrict__ xdbl,
                       float* __restrict__ u_io, const float* __restrict__ A_log,
                       const float* __restrict__ Dp) {
    int d = blockIdx.x * blockDim.x + threadIdx.x;
    int b = blockIdx.y;
    if (d >= D_INNER) return;
    float Ar[D_STATE], h[D_STATE];
    #pragma unroll
    for (int n = 0; n < D_STATE; ++n) {
        Ar[n] = -expf(A_log[(size_t)d * D_STATE + n]);
        h[n] = 0.0f;
    }
    const float Dd = Dp[d];
    for (int l = 0; l < L_SEQ; ++l) {
        size_t row = (size_t)b * L_SEQ + l;
        float dtv = dt[row * D_INNER + d];
        float uv  = u_io[row * D_INNER + d];
        const float* Bp = xdbl + row * 96 + DT_RANK;       // cols 64..79
        const float* Cp = Bp + D_STATE;                    // cols 80..95
        float y = 0.0f;
        #pragma unroll
        for (int n = 0; n < D_STATE; ++n) {
            float dA = expf(dtv * Ar[n]);
            h[n] = dA * h[n] + dtv * Bp[n] * uv;
            y += h[n] * Cp[n];
        }
        u_io[row * D_INNER + d] = y + uv * Dd;
    }
}

// ---------------------------------------------------------------- gate: y * silu(res) -> bf16
__global__ void k_gate(const float* __restrict__ xr, const float* __restrict__ y32,
                       bf16* __restrict__ y16) {
    long i = (long)blockIdx.x * blockDim.x + threadIdx.x;
    if (i >= (long)M_ROWS * D_INNER) return;
    int d = (int)(i % D_INNER);
    size_t row = (size_t)(i / D_INNER);
    float r = xr[row * (2 * D_INNER) + D_INNER + d];
    y16[i] = f2bf(y32[i] * (r * sigmoidf_(r)));
}

// ---------------------------------------------------------------- host launcher
extern "C" void kernel_launch(void* const* d_in, const int* in_sizes, int n_in,
                              void* d_out, int out_size, void* d_ws, size_t ws_size,
                              hipStream_t stream) {
    (void)in_sizes; (void)n_in; (void)out_size; (void)ws_size;
    const float* x       = (const float*)d_in[0];
    const float* W_in    = (const float*)d_in[1];
    const float* conv_w  = (const float*)d_in[2];
    const float* conv_b  = (const float*)d_in[3];
    const float* W_xproj = (const float*)d_in[4];
    const float* W_dt    = (const float*)d_in[5];
    const float* b_dt    = (const float*)d_in[6];
    const float* A_log   = (const float*)d_in[7];
    const float* Dp      = (const float*)d_in[8];
    const float* W_out   = (const float*)d_in[9];
    float* out = (float*)d_out;

    char* ws = (char*)d_ws;
    size_t cur = 0;
    auto take = [&](size_t bytes) -> void* {
        void* p = ws + cur;
        cur += (bytes + 255) & ~(size_t)255;
        return p;
    };

    float* xr    = (float*)take((size_t)M_ROWS * 2 * D_INNER * 4);   // 64 MB (u_pre | res)
    float* u32   = (float*)take((size_t)M_ROWS * D_INNER * 4);       // 32 MB (u -> y in place)
    bf16*  u16   = (bf16*) take((size_t)M_ROWS * D_INNER * 2);       // 16 MB (u bf16 -> yg bf16)
    float* xdbl  = (float*)take((size_t)M_ROWS * 96 * 4);            // 1.5 MB
    float* dtb   = (float*)take((size_t)M_ROWS * D_INNER * 4);       // 32 MB
    bf16*  xb    = (bf16*) take((size_t)M_ROWS * D_MODEL * 2);       // 8 MB
    bf16*  wint  = (bf16*) take((size_t)(2 * D_INNER) * D_MODEL * 2);// 8 MB  [4096 x 1024]
    bf16*  wxpt  = (bf16*) take((size_t)96 * D_INNER * 2);           //      [96 x 2048]
    bf16*  dtl16 = (bf16*) take((size_t)M_ROWS * DT_RANK * 2);       //      [4096 x 64]
    bf16*  wdtt  = (bf16*) take((size_t)D_INNER * DT_RANK * 2);      //      [2048 x 64]
    bf16*  woutt = (bf16*) take((size_t)D_MODEL * D_INNER * 2);      // 4 MB [1024 x 2048]

    const int T = 256;
    auto nb = [](long n, int t) { return (unsigned)((n + t - 1) / t); };

    // Precision prep: activations + weights -> bf16 (weights transposed to N x K)
    k_cvt_bf16<<<nb((long)M_ROWS * D_MODEL, T), T, 0, stream>>>(x, xb, (long)M_ROWS * D_MODEL);
    k_transpose_bf16<<<nb((long)D_MODEL * 2 * D_INNER, T), T, 0, stream>>>(W_in, wint, D_MODEL, 2 * D_INNER);
    k_transpose_bf16<<<nb((long)D_INNER * 96, T), T, 0, stream>>>(W_xproj, wxpt, D_INNER, 96);
    k_transpose_bf16<<<nb((long)DT_RANK * D_INNER, T), T, 0, stream>>>(W_dt, wdtt, DT_RANK, D_INNER);
    k_transpose_bf16<<<nb((long)D_INNER * D_MODEL, T), T, 0, stream>>>(W_out, woutt, D_INNER, D_MODEL);

    // GEMM 1: xr = x @ W_in   (4096 x 4096, K=1024) — blocked + async-LDS
    {
        dim3 g((2 * D_INNER) / 128, M_ROWS / 128);
        k_gemm_bf16_blk<<<g, 256, 0, stream>>>(xb, wint, xr, M_ROWS, 2 * D_INNER, D_MODEL);
    }
    // depthwise causal conv + bias + SiLU
    k_conv_silu<<<nb((long)M_ROWS * D_INNER, T), T, 0, stream>>>(xr, conv_w, conv_b, u32, u16);
    // GEMM 2: x_dbl = u @ W_xproj   (4096 x 96, K=2048) — skinny N, simple kernel
    {
        long tiles = (long)(M_ROWS / 16) * (96 / 16);
        k_gemm_bf16<<<nb(tiles, 8), 256, 0, stream>>>(u16, wxpt, xdbl, M_ROWS, 96, D_INNER);
    }
    k_cvt_dtlow<<<nb((long)M_ROWS * DT_RANK, T), T, 0, stream>>>(xdbl, dtl16);
    // GEMM 3: dt_lin = dt_low @ W_dt   (4096 x 2048, K=64) — blocked + async-LDS
    {
        dim3 g(D_INNER / 128, M_ROWS / 128);
        k_gemm_bf16_blk<<<g, 256, 0, stream>>>(dtl16, wdtt, dtb, M_ROWS, D_INNER, DT_RANK);
    }
    k_softplus<<<nb((long)M_ROWS * D_INNER, T), T, 0, stream>>>(dtb, b_dt);
    // selective scan: 4096 channels, serial over L
    {
        dim3 g(D_INNER / 256, B_SZ);
        k_scan<<<g, 256, 0, stream>>>(dtb, xdbl, u32, A_log, Dp);
    }
    // gate with silu(res) -> bf16 (reuses u16)
    k_gate<<<nb((long)M_ROWS * D_INNER, T), T, 0, stream>>>(xr, u32, u16);
    // GEMM 4: out = yg @ W_out   (4096 x 1024, K=2048) — blocked + async-LDS
    {
        dim3 g(D_MODEL / 128, M_ROWS / 128);
        k_gemm_bf16_blk<<<g, 256, 0, stream>>>(u16, woutt, out, M_ROWS, D_MODEL, D_INNER);
    }
}
